// SphereConv2D_16887811407989
// MI455X (gfx1250) — matrix operand order repaired and checked by reading, not compiled, not run on hardware
//
#include <hip/hip_runtime.h>
#include <math.h>

#define H_IN  128
#define W_IN  256
#define CIN   64
#define COUT  128
#define NB    4

typedef float  v2f  __attribute__((ext_vector_type(2)));
typedef float  v4f  __attribute__((ext_vector_type(4)));
typedef float  v8f  __attribute__((ext_vector_type(8)));
typedef __bf16 bf4  __attribute__((ext_vector_type(4)));
typedef __bf16 v16bf __attribute__((ext_vector_type(16)));

// workspace layout (float offsets)
//   xT     : [N][H][W][CIN]       = 8,388,608 floats (NHWC transpose of x)
//   coords : [H][W][9][2]         =   589,824 floats (ix, iy per tap)
//   wswzH/L: bf16, fragment-swizzled weights, 73,728 elements each
#define WS_XT   0
#define WS_CO   (NB * H_IN * W_IN * CIN)           // 8388608
#define WS_WB   (WS_CO + H_IN * W_IN * 9 * 2)      // 8978432 (float offset)
#define NWELEM  (9 * CIN * COUT)                   // 73728

__device__ __forceinline__ int iclampi(int v, int lo, int hi) {
  return v < lo ? lo : (v > hi ? hi : v);
}

// ---------------------------------------------------------------------------
// Kernel 1: NCHW -> NHWC transpose (LDS tiled, coalesced both sides)
// ---------------------------------------------------------------------------
__global__ __launch_bounds__(256) void k_transpose(const float* __restrict__ x,
                                                   float* __restrict__ xT) {
  __shared__ float tile[64][65];
  const int b  = blockIdx.x;
  const int xc = b & 3;            // W/64 = 4 chunks
  const int ny = b >> 2;           // n*H + y
  const int nn = ny / H_IN;
  const int yy = ny % H_IN;
  const int x0 = xc * 64;
  const int tx = threadIdx.x & 63;
  const int ty = threadIdx.x >> 6; // 0..3

#pragma unroll
  for (int i = 0; i < 16; ++i) {
    const int ci = ty * 16 + i;
    tile[ci][tx] = x[(((size_t)nn * CIN + ci) * H_IN + yy) * W_IN + x0 + tx];
  }
  __syncthreads();
#pragma unroll
  for (int i = 0; i < 16; ++i) {
    const int xx = ty * 16 + i;
    xT[(((size_t)ny) * W_IN + (x0 + xx)) * CIN + tx] = tile[tx][xx];
  }
}

// ---------------------------------------------------------------------------
// Kernel 2: spherical grid -> bilinear sample coordinates (double precision,
// matching the float64 numpy reference). One thread per (r, c, tap).
// ---------------------------------------------------------------------------
__global__ __launch_bounds__(256) void k_coords(float* __restrict__ coords) {
  const int gid = blockIdx.x * 256 + threadIdx.x;
  if (gid >= H_IN * W_IN * 9) return;
  const int t  = gid % 9;
  const int rc = gid / 9;
  const int c  = rc % W_IN;
  const int r  = rc / W_IN;
  const int kh = t / 3, kw = t % 3;

  const double pi = 3.14159265358979323846;
  double nr, nc;
  if (kh == 1 && kw == 1) {
    nr = (double)r;
    nc = (double)c;
  } else {
    const double dphi = pi / H_IN;
    const double dth  = 2.0 * pi / W_IN;
    const double tt = tan(dth);
    const double pp = tan(dphi);
    const double sp = pp / cos(dth);
    const double xrow[3] = { -tt, 0.0, tt };
    double X = xrow[kw];
    double Y;
    if (kh == 0)      Y = (kw == 1) ? pp : sp;
    else if (kh == 1) Y = 0.0;
    else              Y = (kw == 1) ? -pp : -sp;

    const double rho = sqrt(X * X + Y * Y);
    const double v   = atan(rho);
    const double phi   = -(((double)r + 0.5) / H_IN * pi - pi / 2.0);
    const double theta = ((double)c + 0.5) / W_IN * 2.0 * pi - pi;
    const double sv = sin(v), cv = cos(v);
    const double sphi = sin(phi), cphi = cos(phi);
    double arg = cv * sphi + Y * sv * cphi / rho;
    arg = fmin(1.0, fmax(-1.0, arg));
    const double nphi   = asin(arg);
    const double ntheta = theta + atan(X * sv / (rho * cphi * cv - Y * sphi * sv));
    nr = (-nphi + pi / 2.0) * H_IN / pi - 0.5;
    nc = (ntheta + pi) * W_IN / (2.0 * pi) - 0.5;
    nc = fmod(nc + (double)W_IN, (double)W_IN);
  }
  const double gx = nc * 2.0 / W_IN - 1.0;
  const double gy = nr * 2.0 / H_IN - 1.0;
  coords[gid * 2 + 0] = (float)(((gx + 1.0) * W_IN - 1.0) * 0.5); // ix
  coords[gid * 2 + 1] = (float)(((gy + 1.0) * H_IN - 1.0) * 0.5); // iy
}

// ---------------------------------------------------------------------------
// Kernel 3: weight split + swizzle.
// Output layout = per-lane A-fragment order this file uses for V_WMMA_F32_16X16X32_BF16 (contiguous k):
//   entry(t, cb, co, hi) = 16 bf16 = lane's fragment half for K-chunk cb.
//   hi=0 lane gets K{0..15}, hi=1 lane gets K{16..31} (+cb*32).
//   gid = t*8192 + cb*4096 + co*32 + hi*16 + klane
// ---------------------------------------------------------------------------
__global__ __launch_bounds__(256) void k_wswz(const float* __restrict__ w,
                                              __bf16* __restrict__ wh,
                                              __bf16* __restrict__ wl) {
  const int gid = blockIdx.x * 256 + threadIdx.x;
  if (gid >= NWELEM) return;
  const int klane = gid & 15;
  const int hi    = (gid >> 4) & 1;
  const int co    = (gid >> 5) & 127;
  const int cb    = (gid >> 12) & 1;
  const int t     = gid >> 13;
  const int k8 = hi * 16 + klane;
  const int ci = cb * 32 + k8;
  const float v = w[((size_t)co * CIN + ci) * 9 + t];
  const __bf16 h = (__bf16)v;
  wh[gid] = h;
  wl[gid] = (__bf16)(v - (float)h);
}

// ---------------------------------------------------------------------------
// Kernel 4: fused bilinear-sample + implicit GEMM, split-precision bf16:
//   C = (A_hi+A_lo)(B_hi+B_lo) ~= A_hi*B_hi + A_hi*B_lo + A_lo*B_hi
// accumulated in f32 via V_WMMA_F32_16X16X32_BF16 (3 WMMAs per K=32 chunk,
// vs 8 f32 K=4 WMMAs -> ~2.7x fewer matrix-pipe issue slots, ~fp32 accuracy).
// One WG (256 threads / 8 waves) computes a 128(co) x 128(pos) output tile.
// ---------------------------------------------------------------------------
__global__ __launch_bounds__(256) void k_main(const float* __restrict__ xT,
                                              const float* __restrict__ coords,
                                              const __bf16* __restrict__ wswzH,
                                              const __bf16* __restrict__ wswzL,
                                              const float* __restrict__ bias,
                                              float* __restrict__ out) {
  __shared__ __bf16 S_hi[128][72];  // samples hi plane [pos][k], pad 72
  __shared__ __bf16 S_lo[128][72];  // samples lo plane

  const int tid  = threadIdx.x;
  const int lane = tid & 31;
  const int wv   = tid >> 5;     // wave 0..7 -> co rows [wv*16, wv*16+16)
  const int hi   = lane >> 4;    // K-half selector
  const int ln   = lane & 15;

  const int m0 = blockIdx.x * 128;        // global position base
  const int n  = m0 >> 15;                // / (H*W)
  const int rc = m0 & (H_IN * W_IN - 1);
  const int r  = rc >> 8;                 // / W
  const int c0 = rc & (W_IN - 1);         // 0 or 128

  v8f acc[8];
#pragma unroll
  for (int j = 0; j < 8; ++j)
#pragma unroll
    for (int e = 0; e < 8; ++e) acc[j][e] = 0.0f;

  // sampling work split: thread pair per position, 32 channels each
  const int p     = tid >> 1;
  const int halfc = (tid & 1) * 32;
  const int cbase = ((r * W_IN + c0 + p) * 9) * 2;

  for (int t = 0; t < 9; ++t) {
    __syncthreads();

    // ---- bilinear sample 128 positions x 64 channels, split into hi/lo ----
    const float ix = coords[cbase + t * 2 + 0];
    const float iy = coords[cbase + t * 2 + 1];
    const float x0f = floorf(ix), y0f = floorf(iy);
    const float fx = ix - x0f, fy = iy - y0f;
    const int xi0 = (int)x0f, yi0 = (int)y0f;
    const int xi1 = xi0 + 1,  yi1 = yi0 + 1;
    const float vx0 = (xi0 >= 0 && xi0 < W_IN) ? 1.f : 0.f;
    const float vx1 = (xi1 >= 0 && xi1 < W_IN) ? 1.f : 0.f;
    const float vy0 = (yi0 >= 0 && yi0 < H_IN) ? 1.f : 0.f;
    const float vy1 = (yi1 >= 0 && yi1 < H_IN) ? 1.f : 0.f;
    const float w00 = (1.f - fy) * (1.f - fx) * vy0 * vx0;
    const float w01 = (1.f - fy) * fx         * vy0 * vx1;
    const float w10 = fy         * (1.f - fx) * vy1 * vx0;
    const float w11 = fy         * fx         * vy1 * vx1;
    const int cx0 = iclampi(xi0, 0, W_IN - 1), cx1 = iclampi(xi1, 0, W_IN - 1);
    const int cy0 = iclampi(yi0, 0, H_IN - 1), cy1 = iclampi(yi1, 0, H_IN - 1);
    const float* b00 = xT + (((size_t)n * H_IN + cy0) * W_IN + cx0) * CIN + halfc;
    const float* b01 = xT + (((size_t)n * H_IN + cy0) * W_IN + cx1) * CIN + halfc;
    const float* b10 = xT + (((size_t)n * H_IN + cy1) * W_IN + cx0) * CIN + halfc;
    const float* b11 = xT + (((size_t)n * H_IN + cy1) * W_IN + cx1) * CIN + halfc;
#pragma unroll
    for (int q = 0; q < 8; ++q) {
      v4f s = w00 * *(const v4f*)(b00 + q * 4)
            + w01 * *(const v4f*)(b01 + q * 4)
            + w10 * *(const v4f*)(b10 + q * 4)
            + w11 * *(const v4f*)(b11 + q * 4);
      bf4 sh, sl;
#pragma unroll
      for (int e = 0; e < 4; ++e) {
        const float f = s[e];
        const __bf16 h = (__bf16)f;
        sh[e] = h;
        sl[e] = (__bf16)(f - (float)h);
      }
      *(bf4*)&S_hi[p][halfc + q * 4] = sh;
      *(bf4*)&S_lo[p][halfc + q * 4] = sl;
    }
    __syncthreads();

    // ---- preload this tap's weight fragments (pre-swizzled, 32B/lane) ----
    v16bf aH[2], aL[2];
#pragma unroll
    for (int cb = 0; cb < 2; ++cb) {
      const size_t fo = (((size_t)(t * 2 + cb) * COUT) + (wv * 16 + ln)) * 32 + hi * 16;
      aH[cb] = *(const v16bf*)(wswzH + fo);
      aL[cb] = *(const v16bf*)(wswzL + fo);
    }

    // ---- 2 K-chunks x 8 pos tiles x 3 split-products of bf16 WMMA ----
#pragma unroll
    for (int cb = 0; cb < 2; ++cb) {
#pragma unroll
      for (int j = 0; j < 8; ++j) {
        const v16bf bH = *(const v16bf*)&S_hi[j * 16 + ln][cb * 32 + hi * 16];
        const v16bf bL = *(const v16bf*)&S_lo[j * 16 + ln][cb * 32 + hi * 16];
        acc[j] = __builtin_amdgcn_wmma_f32_16x16x32_bf16(
            false, aH[cb], false, bH, (short)0, acc[j], false, false);
        acc[j] = __builtin_amdgcn_wmma_f32_16x16x32_bf16(
            false, aH[cb], false, bL, (short)0, acc[j], false, false);
        acc[j] = __builtin_amdgcn_wmma_f32_16x16x32_bf16(
            false, aL[cb], false, bH, (short)0, acc[j], false, false);
      }
    }
  }

  // ---- epilogue: bias + store (half-wave contiguous along W) ----
#pragma unroll
  for (int vr = 0; vr < 8; ++vr) {
    const int co = wv * 16 + vr + hi * 8;
    const float bv = bias[co];
    const size_t obase = (((size_t)n * COUT + co) * H_IN + r) * W_IN + c0 + ln;
#pragma unroll
    for (int j = 0; j < 8; ++j) {
      out[obase + (size_t)j * 16] = acc[j][vr] + bv;
    }
  }
}

// ---------------------------------------------------------------------------
extern "C" void kernel_launch(void* const* d_in, const int* in_sizes, int n_in,
                              void* d_out, int out_size, void* d_ws, size_t ws_size,
                              hipStream_t stream) {
  const float* x    = (const float*)d_in[0];
  const float* w    = (const float*)d_in[1];
  const float* bias = (const float*)d_in[2];
  float* ws      = (float*)d_ws;
  float* xT      = ws + WS_XT;
  float* coords  = ws + WS_CO;
  __bf16* wswzH  = (__bf16*)(ws + WS_WB);
  __bf16* wswzL  = wswzH + NWELEM;
  float* out     = (float*)d_out;

  k_transpose<<<NB * H_IN * (W_IN / 64), 256, 0, stream>>>(x, xT);
  k_coords<<<(H_IN * W_IN * 9 + 255) / 256, 256, 0, stream>>>(coords);
  k_wswz<<<(NWELEM + 255) / 256, 256, 0, stream>>>(w, wswzH, wswzL);
  k_main<<<(NB * H_IN * W_IN) / 128, 256, 0, stream>>>(xT, coords, wswzH, wswzL, bias, out);
}
